// GroupMambaLayer_4999341932553
// MI455X (gfx1250) — compile-verified, matrix operand order checked by reading, not run on hardware
//
#include <hip/hip_runtime.h>
#include <hip/hip_bf16.h>

typedef __attribute__((ext_vector_type(16))) _Float16 v16h;
typedef __attribute__((ext_vector_type(8)))  float    v8f;

// ---- helpers -------------------------------------------------------------

__device__ inline v16h ld16_f32h(const float* p) {
    v16h r;
#pragma unroll
    for (int i = 0; i < 16; ++i) r[i] = (_Float16)p[i];
    return r;
}

__device__ inline float silu_f(float v) { return v / (1.0f + expf(-v)); }

// ---- K1 / K9a: layernorm over C=256, templated output type ---------------

template <typename OUT>
__global__ void k_ln256(const float* __restrict__ x, const float* __restrict__ g,
                        const float* __restrict__ b, OUT* __restrict__ out) {
    __shared__ float red[256];
    const size_t tok = blockIdx.x;
    const int c = threadIdx.x;
    float v = x[tok * 256 + c];
    red[c] = v;
    __syncthreads();
    for (int s = 128; s > 0; s >>= 1) {
        if (c < s) red[c] += red[c + s];
        __syncthreads();
    }
    float mean = red[0] * (1.0f / 256.0f);
    __syncthreads();
    float d = v - mean;
    red[c] = d * d;
    __syncthreads();
    for (int s = 128; s > 0; s >>= 1) {
        if (c < s) red[c] += red[c + s];
        __syncthreads();
    }
    float var = red[0] * (1.0f / 256.0f);
    float r = d * rsqrtf(var + 1e-5f) * g[c] + b[c];
    out[tok * 256 + c] = (OUT)r;
}

// ---- K2: z_avg = mean over tokens (two stage, no atomics) ----------------

__global__ void k_zavg_part(const float* __restrict__ xln, float* __restrict__ part) {
    const int chunk = blockIdx.x;   // 0..15
    const int b = blockIdx.y;
    const int c = threadIdx.x;      // 0..255
    float s = 0.0f;
    const size_t base = ((size_t)b * 4096 + (size_t)chunk * 256) * 256 + c;
    for (int i = 0; i < 256; ++i) s += xln[base + (size_t)i * 256];
    part[((size_t)b * 16 + chunk) * 256 + c] = s;
}

__global__ void k_zavg_final(const float* __restrict__ part, float* __restrict__ zavg) {
    const int b = blockIdx.x;
    const int c = threadIdx.x;
    float s = 0.0f;
    for (int k = 0; k < 16; ++k) s += part[((size_t)b * 16 + k) * 256 + c];
    zavg[(size_t)b * 256 + c] = s * (1.0f / 4096.0f);
}

// ---- K3: gate = sigmoid(z_avg @ fc_w.T + fc_b + conv1d(z_avg)) -----------

__global__ void k_gate(const float* __restrict__ zavg, const float* __restrict__ fcw,
                       const float* __restrict__ fcb, const float* __restrict__ cw,
                       float* __restrict__ gate) {
    const int b = blockIdx.x;
    const int i = threadIdx.x;     // 0..255
    const float* z = zavg + (size_t)b * 256;
    float conv = 0.0f;
#pragma unroll
    for (int k = 0; k < 3; ++k) {
        int j = i + k - 1;
        if (j >= 0 && j < 256) conv += cw[k] * z[j];
    }
    float dot = 0.0f;
    for (int k = 0; k < 256; ++k) dot += z[k] * fcw[(size_t)i * 256 + k];
    float v = dot + fcb[i] + conv;
    gate[(size_t)b * 256 + i] = 1.0f / (1.0f + expf(-v));
}

// ---- K4: in_proj GEMM (WMMA f16): [xx | z] = xln_part @ in_w.T -----------
// M = B*4096, K = 64, N = 128. One 16x16 tile per wave, 8 waves/block.

__global__ void k_inproj(const float* __restrict__ xln, const float* __restrict__ w,
                         float* __restrict__ xx, float* __restrict__ zb, int B) {
    const int g = blockIdx.y;
    const int mt = blockIdx.x;
    const int wv = threadIdx.x >> 5;       // N-tile 0..7
    const int lane = threadIdx.x & 31;
    const int row = mt * 16 + (lane & 15);
    const int kh = (lane >> 4) * 16;
    const int n = wv * 16 + (lane & 15);
    const float* ap = xln + (size_t)row * 256 + g * 64 + kh;
    const float* bp = w + ((size_t)(g * 128 + n)) * 64 + kh;
    v8f acc = {};
#pragma unroll
    for (int k0 = 0; k0 < 64; k0 += 32) {
        v16h a = ld16_f32h(ap + k0);
        v16h bm = ld16_f32h(bp + k0);
        acc = __builtin_amdgcn_wmma_f32_16x16x32_f16(false, a, false, bm,
                                                     (short)0, acc, false, false);
    }
    const int col = wv * 16 + (lane & 15);
#pragma unroll
    for (int j = 0; j < 8; ++j) {
        const int m = mt * 16 + j + ((lane >> 4) << 3);
        const int b = m >> 12;
        const int l = m & 4095;
        const float v = acc[j];
        const size_t gb = (size_t)g * B + b;
        if (col < 64)
            xx[(gb * 64 + col) * 4096 + l] = v;
        else
            zb[(gb * 4096 + l) * 64 + (col - 64)] = v;
    }
}

// ---- K5: depthwise 3x3 conv + bias + silu + to_seq reorder ---------------

__global__ void k_dwconv(const float* __restrict__ xx, const float* __restrict__ cw,
                         const float* __restrict__ cb, float* __restrict__ sb, int B) {
    const size_t t = (size_t)blockIdx.x * 256 + threadIdx.x;
    const int w = (int)(t & 63);
    const int h = (int)((t >> 6) & 63);
    const int ch = (int)((t >> 12) & 63);
    const int gb = (int)(t >> 18);         // g*B + b
    const int g = gb / B;
    const float* in = xx + ((size_t)gb * 64 + ch) * 4096;
    const float* wk = cw + ((size_t)g * 64 + ch) * 9;
    float acc = 0.0f;
#pragma unroll
    for (int dy = -1; dy <= 1; ++dy)
#pragma unroll
        for (int dx = -1; dx <= 1; ++dx) {
            const int hh = h + dy, ww = w + dx;
            if (hh >= 0 && hh < 64 && ww >= 0 && ww < 64)
                acc += wk[(dy + 1) * 3 + (dx + 1)] * in[hh * 64 + ww];
        }
    acc += cb[g * 64 + ch];
    acc = silu_f(acc);
    int lp;
    if (g == 0)      lp = h * 64 + w;
    else if (g == 1) lp = w * 64 + h;
    else if (g == 2) lp = 4095 - (h * 64 + w);
    else             lp = 4095 - (w * 64 + h);
    sb[((size_t)gb * 64 + ch) * 4096 + lp] = acc;
}

// ---- K6: x_dbl = xp_w @ s ; dt = softplus(dt_w @ dt_r + dt_b) ------------

__global__ void k_xdbl(const float* __restrict__ sb, const float* __restrict__ xpw,
                       const float* __restrict__ dtw, const float* __restrict__ dtbias,
                       float* __restrict__ dtb, float* __restrict__ bsb,
                       float* __restrict__ csb, int B) {
    const int gb = blockIdx.y;             // g*B + b
    const int g = gb / B;
    const int l = blockIdx.x * 256 + threadIdx.x;
    const float* s = sb + (size_t)gb * 64 * 4096;
    float xd[6] = {0, 0, 0, 0, 0, 0};
    for (int c = 0; c < 64; ++c) {
        const float sv = s[(size_t)c * 4096 + l];
#pragma unroll
        for (int r = 0; r < 6; ++r) xd[r] += xpw[((size_t)g * 6 + r) * 64 + c] * sv;
    }
    bsb[(size_t)gb * 4096 + l] = xd[4];
    csb[(size_t)gb * 4096 + l] = xd[5];
    for (int d = 0; d < 64; ++d) {
        const float* dw = dtw + ((size_t)g * 64 + d) * 4;
        float v = dw[0] * xd[0] + dw[1] * xd[1] + dw[2] * xd[2] + dw[3] * xd[3]
                + dtbias[g * 64 + d];
        float sp = (v > 20.0f) ? v : log1pf(expf(v));
        dtb[((size_t)gb * 64 + d) * 4096 + l] = sp;
    }
}

// ---- K7: selective scan (n=1), wave-parallel chunked linear recurrence ---
// One wave per (g,b,ch); lane j owns segment [j*128, j*128+128). Composition
// of affine ops (h -> a*h + b) scanned across lanes via shuffles.

__global__ void k_scan(float* __restrict__ sb /* u in, y out in-place */,
                       const float* __restrict__ dtb, const float* __restrict__ bsb,
                       const float* __restrict__ csb, const float* __restrict__ Alog,
                       const float* __restrict__ Dp, int B) {
    const int wid = (int)(((size_t)blockIdx.x * 256 + threadIdx.x) >> 5);
    const int lane = threadIdx.x & 31;
    const int ch = wid & 63;
    const int gb = wid >> 6;
    const int g = gb / B;
    const float A = -expf(Alog[g * 64 + ch]);
    const float Dv = Dp[g * 64 + ch];
    const size_t base = ((size_t)gb * 64 + ch) * 4096;
    const float* dtp = dtb + base;
    float* up = sb + base;
    const float* Bp = bsb + (size_t)gb * 4096;
    const float* Cp = csb + (size_t)gb * 4096;
    const int l0 = lane * 128;

    // pass 1: compose segment op
    float a = 1.0f, bb = 0.0f;
    for (int i = 0; i < 128; ++i) {
        const int l = l0 + i;
        const float dt = dtp[l];
        const float dA = expf(dt * A);
        const float dBu = dt * Bp[l] * up[l];
        bb = dA * bb + dBu;
        a = dA * a;
    }
    // inclusive scan of op composition across lanes
    for (int off = 1; off < 32; off <<= 1) {
        const float au = __shfl_up(a, off, 32);
        const float bu = __shfl_up(bb, off, 32);
        if (lane >= off) {
            bb = a * bu + bb;
            a = a * au;
        }
    }
    float h = __shfl_up(bb, 1, 32);
    if (lane == 0) h = 0.0f;

    // pass 2: rescan with correct incoming state, emit y in place
    for (int i = 0; i < 128; ++i) {
        const int l = l0 + i;
        const float dt = dtp[l];
        const float u = up[l];
        const float dA = expf(dt * A);
        h = dA * h + dt * Bp[l] * u;
        up[l] = h * Cp[l] + Dv * u;
    }
}

// ---- K8a: from_seq gather + LN(c=64) + *silu(z) -> t (f16) ---------------

__global__ void k_ynorm(const float* __restrict__ yb, const float* __restrict__ zb,
                        const float* __restrict__ ong, const float* __restrict__ onb,
                        _Float16* __restrict__ tb, int B) {
    const int wtok = (int)(((size_t)blockIdx.x * 256 + threadIdx.x) >> 5);
    const int lane = threadIdx.x & 31;
    const int l = wtok & 4095;
    const int gb = wtok >> 12;
    const int g = gb / B;
    const int h = l >> 6, w = l & 63;
    int lp;
    if (g == 0)      lp = l;
    else if (g == 1) lp = w * 64 + h;
    else if (g == 2) lp = 4095 - l;
    else             lp = 4095 - (w * 64 + h);
    const size_t ybase = (size_t)gb * 64 * 4096;
    const float v0 = yb[ybase + (size_t)lane * 4096 + lp];
    const float v1 = yb[ybase + (size_t)(lane + 32) * 4096 + lp];
    float s = v0 + v1;
    for (int off = 16; off > 0; off >>= 1) s += __shfl_xor(s, off, 32);
    const float mean = s * (1.0f / 64.0f);
    const float d0 = v0 - mean, d1 = v1 - mean;
    float q = d0 * d0 + d1 * d1;
    for (int off = 16; off > 0; off >>= 1) q += __shfl_xor(q, off, 32);
    const float rstd = rsqrtf(q * (1.0f / 64.0f) + 1e-5f);
    const size_t zbase = ((size_t)gb * 4096 + l) * 64;
    const float o0 = (d0 * rstd * ong[g * 64 + lane] + onb[g * 64 + lane])
                   * silu_f(zb[zbase + lane]);
    const float o1 = (d1 * rstd * ong[g * 64 + lane + 32] + onb[g * 64 + lane + 32])
                   * silu_f(zb[zbase + lane + 32]);
    tb[zbase + lane] = (_Float16)o0;
    tb[zbase + lane + 32] = (_Float16)o1;
}

// ---- K8b: out_proj GEMM (WMMA) fused with *skip*xg*gate -> xm ------------
// M = B*4096, K = 64, N = 64. 8 waves/block = 2 M-tiles x 4 N-tiles.

__global__ void k_outproj(const _Float16* __restrict__ tb, const float* __restrict__ wout,
                          const float* __restrict__ xln, const float* __restrict__ gate,
                          const float* __restrict__ skip, float* __restrict__ xm, int B) {
    const int g = blockIdx.y;
    const int wv = threadIdx.x >> 5;
    const int lane = threadIdx.x & 31;
    const int mt = blockIdx.x * 2 + (wv >> 2);
    const int nt = wv & 3;
    const int row = mt * 16 + (lane & 15);
    const int kh = (lane >> 4) * 16;
    const int n = nt * 16 + (lane & 15);
    const _Float16* ap = tb + (size_t)g * B * 4096 * 64 + (size_t)row * 64 + kh;
    const float* bp = wout + ((size_t)g * 64 + n) * 64 + kh;
    v8f acc = {};
#pragma unroll
    for (int k0 = 0; k0 < 64; k0 += 32) {
        v16h a = *(const v16h*)(ap + k0);
        v16h bm = ld16_f32h(bp + k0);
        acc = __builtin_amdgcn_wmma_f32_16x16x32_f16(false, a, false, bm,
                                                     (short)0, acc, false, false);
    }
    const float sk = skip[0];
    const int cg = g * 64 + nt * 16 + (lane & 15);
#pragma unroll
    for (int j = 0; j < 8; ++j) {
        const int m = mt * 16 + j + ((lane >> 4) << 3);
        const int b = m >> 12;
        const float v = acc[j] * sk * xln[(size_t)m * 256 + cg] * gate[(size_t)b * 256 + cg];
        xm[(size_t)m * 256 + cg] = v;
    }
}

// ---- K9b: final GEMM (WMMA) out = xm_ln @ proj_w.T + proj_b --------------
// M = B*4096, K = 256, N = 256.

__global__ void k_finalproj(const _Float16* __restrict__ xmln, const float* __restrict__ pw,
                            const float* __restrict__ pb, float* __restrict__ out) {
    const int wv = threadIdx.x >> 5;
    const int lane = threadIdx.x & 31;
    const int mt = blockIdx.x;
    const int nt = blockIdx.y * 8 + wv;
    const int row = mt * 16 + (lane & 15);
    const int kh = (lane >> 4) * 16;
    const int n = nt * 16 + (lane & 15);
    const _Float16* ap = xmln + (size_t)row * 256 + kh;
    const float* bp = pw + (size_t)n * 256 + kh;
    v8f acc = {};
#pragma unroll
    for (int k0 = 0; k0 < 256; k0 += 32) {
        v16h a = *(const v16h*)(ap + k0);
        v16h bm = ld16_f32h(bp + k0);
        acc = __builtin_amdgcn_wmma_f32_16x16x32_f16(false, a, false, bm,
                                                     (short)0, acc, false, false);
    }
    const int col = nt * 16 + (lane & 15);
    const float bias = pb[col];
#pragma unroll
    for (int j = 0; j < 8; ++j) {
        const int m = mt * 16 + j + ((lane >> 4) << 3);
        out[(size_t)m * 256 + col] = acc[j] + bias;
    }
}

// ---- host launcher -------------------------------------------------------

extern "C" void kernel_launch(void* const* d_in, const int* in_sizes, int n_in,
                              void* d_out, int out_size, void* d_ws, size_t ws_size,
                              hipStream_t stream) {
    (void)n_in; (void)out_size; (void)ws_size;
    const float* x      = (const float*)d_in[0];
    const float* norm_g = (const float*)d_in[3];
    const float* norm_b = (const float*)d_in[4];
    const float* fc_w   = (const float*)d_in[5];
    const float* fc_b   = (const float*)d_in[6];
    const float* c1w    = (const float*)d_in[7];
    const float* proj_w = (const float*)d_in[8];
    const float* proj_b = (const float*)d_in[9];
    const float* skip   = (const float*)d_in[10];
    const float* in_w   = (const float*)d_in[11];
    const float* c2w    = (const float*)d_in[12];
    const float* c2b    = (const float*)d_in[13];
    const float* xpw    = (const float*)d_in[14];
    const float* dtw    = (const float*)d_in[15];
    const float* dtbias = (const float*)d_in[16];
    const float* Alog   = (const float*)d_in[17];
    const float* Dpv    = (const float*)d_in[18];
    const float* ong    = (const float*)d_in[19];
    const float* onb    = (const float*)d_in[20];
    const float* outw   = (const float*)d_in[21];
    float* out = (float*)d_out;

    const int B = in_sizes[0] / (4096 * 256);
    const size_t BN = (size_t)B * 4096;

    char* ws = (char*)d_ws;
    size_t off = 0;
    auto alloc = [&](size_t bytes) -> void* {
        void* p = ws + off;
        off = (off + bytes + 255) & ~(size_t)255;
        return p;
    };
    float*    xln   = (float*)alloc(BN * 256 * 4);
    float*    part  = (float*)alloc((size_t)B * 16 * 256 * 4);
    float*    zavg  = (float*)alloc((size_t)B * 256 * 4);
    float*    gate  = (float*)alloc((size_t)B * 256 * 4);
    float*    xx    = (float*)alloc((size_t)4 * B * 64 * 4096 * 4);
    float*    zbuf  = (float*)alloc((size_t)4 * B * 4096 * 64 * 4);
    float*    sbuf  = (float*)alloc((size_t)4 * B * 64 * 4096 * 4);  // s, then y in place
    float*    dtbuf = (float*)alloc((size_t)4 * B * 64 * 4096 * 4);
    float*    bsb   = (float*)alloc((size_t)4 * B * 4096 * 4);
    float*    csb   = (float*)alloc((size_t)4 * B * 4096 * 4);
    _Float16* tbuf  = (_Float16*)alloc((size_t)4 * B * 4096 * 64 * 2);
    float*    xm    = (float*)alloc(BN * 256 * 4);
    _Float16* xmln  = (_Float16*)alloc(BN * 256 * 2);

    // 1) layernorm(x) -> xln (f32)
    k_ln256<float><<<dim3((unsigned)BN), 256, 0, stream>>>(x, norm_g, norm_b, xln);
    // 2) z_avg
    k_zavg_part<<<dim3(16, B), 256, 0, stream>>>(xln, part);
    k_zavg_final<<<dim3(B), 256, 0, stream>>>(part, zavg);
    // 3) gate
    k_gate<<<dim3(B), 256, 0, stream>>>(zavg, fc_w, fc_b, c1w, gate);
    // 4) in_proj WMMA GEMM -> xx (g,B,c,HW), z (g,B,HW,c)
    k_inproj<<<dim3((unsigned)(BN / 16), 4), 256, 0, stream>>>(xln, in_w, xx, zbuf, B);
    // 5) depthwise conv + silu + to_seq -> sbuf
    k_dwconv<<<dim3((unsigned)(4 * BN * 64 / 256)), 256, 0, stream>>>(xx, c2w, c2b, sbuf, B);
    // 6) x_dbl + dt projection
    k_xdbl<<<dim3(16, 4 * B), 256, 0, stream>>>(sbuf, xpw, dtw, dtbias, dtbuf, bsb, csb, B);
    // 7) selective scan (y overwrites sbuf)
    k_scan<<<dim3((unsigned)(4 * B * 64 / 8)), 256, 0, stream>>>(sbuf, dtbuf, bsb, csb,
                                                                 Alog, Dpv, B);
    // 8a) from_seq + LN(64) + silu(z) -> tbuf (f16)
    k_ynorm<<<dim3((unsigned)(4 * BN / 8)), 256, 0, stream>>>(sbuf, zbuf, ong, onb, tbuf, B);
    // 8b) out_proj WMMA GEMM fused with skip*xg*gate -> xm
    k_outproj<<<dim3((unsigned)(BN / 32), 4), 256, 0, stream>>>(tbuf, outw, xln, gate,
                                                                skip, xm, B);
    // 9a) layernorm(xm) -> xmln (f16)
    k_ln256<_Float16><<<dim3((unsigned)BN), 256, 0, stream>>>(xm, norm_g, norm_b, xmln);
    // 9b) final WMMA GEMM + bias -> out
    k_finalproj<<<dim3((unsigned)(BN / 16), 2), 256, 0, stream>>>(xmln, proj_w, proj_b, out);
}
